// FullAttentionMasked_23699629539831
// MI455X (gfx1250) — matrix operand order repaired and checked by reading, not compiled, not run on hardware
//
#include <hip/hip_runtime.h>
#include <hip/hip_bf16.h>

typedef __attribute__((ext_vector_type(16))) _Float16 v16h;
typedef __attribute__((ext_vector_type(4)))  _Float16 v4h;
typedef __attribute__((ext_vector_type(8)))  float    v8f;
typedef __attribute__((ext_vector_type(4)))  unsigned int v4u;
typedef __attribute__((ext_vector_type(4)))  int      v4i;
typedef __attribute__((ext_vector_type(8)))  int      v8i;

#define B_    2
#define H_    16
#define T_    2048
#define D_    64
#define BK    64          // keys per block iteration
#define BM    64          // queries per workgroup (4 waves x 16)
#define KSTR  72          // LDS row strides (halfs): 144B rows, odd bank stride, 16B aligned
#define VSTR  72
#define PSTR  72

__global__ __launch_bounds__(128) void fa_kernel(
    const float* __restrict__ Q, const float* __restrict__ K,
    const float* __restrict__ V, const unsigned char* __restrict__ Msk,
    float* __restrict__ O)
{
    __shared__ _Float16 Ksm[BK * KSTR];                    // [key][d]  f16
    __shared__ _Float16 Vt [D_ * VSTR];                    // [d][key]  f16 (transposed)
    __shared__ _Float16 Psm[4 * 16 * PSTR];                // per-wave P scratch
    __shared__ __align__(16) unsigned char Msm[BM * BK];   // mask tile (TDM target)

    const int tid  = threadIdx.x;
    const int wave = tid >> 5;
    const int lane = tid & 31;
    const int hl   = lane >> 4;     // half-wave (0/1)
    const int ln   = lane & 15;
    const int q0   = blockIdx.x * BM;
    const int bh   = blockIdx.y;

    const size_t base = (size_t)bh * T_ * D_;
    const float* Qg = Q + base;
    const float* Kg = K + base;
    const float* Vg = V + base;

    const unsigned ldsMsm = (unsigned)(uintptr_t)&Msm[0];  // LDS byte offset (flat[31:0])

    // ---- Q fragments (A layout, f32 -> f16), resident in registers ----
    const int qrow = q0 + wave * 16 + ln;   // A-matrix row M = lane&15
    v16h qa[2];
    {
        const float* qp = Qg + (size_t)qrow * D_;
        #pragma unroll
        for (int f = 0; f < 2; ++f) {        // d offset = 32*f
            #pragma unroll
            for (int i = 0; i < 8; ++i) {
                qa[f][i]     = (_Float16)qp[32 * f +      hl * 8 + i];
                qa[f][i + 8] = (_Float16)qp[32 * f + 16 + hl * 8 + i];
            }
        }
    }

    v8f acc[4] = {};                  // O accumulator, C layout, dt = d/16
    float mrun[8], lrun[8];
    #pragma unroll
    for (int r = 0; r < 8; ++r) { mrun[r] = -3.0e38f; lrun[r] = 0.0f; }

    const float scale = 0.125f;       // 1/sqrt(64)

    for (int k0 = 0; k0 < T_; k0 += BK) {
        __syncthreads();   // all waves done reading Msm/Ksm/Vt from previous block

        // ---- TDM: DMA the 64x64 mask tile (1-byte elems, row stride T) into LDS ----
        if (wave == 0) {
            unsigned long long ma =
                (unsigned long long)(uintptr_t)(Msk + (size_t)q0 * T_ + k0);
            v4u g0 = { 1u,                                   // count=1, user mode
                       ldsMsm,                               // lds_addr
                       (unsigned)ma,                         // global_addr[31:0]
                       (unsigned)(ma >> 32) | (2u << 30) };  // addr[56:32] | type=2
            v8i g1 = { 0,                                    // wg_mask=0, data_size=1B
                       (int)((T_ & 0xffff) << 16),           // tensor_dim0 lo16
                       (int)(((T_ >> 16) & 0xffff) | ((T_ & 0xffff) << 16)), // d0 hi | d1 lo
                       (int)(((T_ >> 16) & 0xffff) | (BK << 16)),            // d1 hi | tile_dim0
                       (int)(BM),                            // tile_dim1=64, tile_dim2=0
                       (int)(T_),                            // tensor_dim0_stride lo32
                       0, 0 };                               // stride hi / dim1_stride
            v4i gz = { 0, 0, 0, 0 };
#if __clang_major__ >= 23
            v8i gz8 = { 0, 0, 0, 0, 0, 0, 0, 0 };
            __builtin_amdgcn_tensor_load_to_lds(g0, g1, gz, gz, gz8, 0);
#else
            __builtin_amdgcn_tensor_load_to_lds(g0, g1, gz, gz, 0);
#endif
        }

        // ---- cooperative K/V fill (float4 loads; K row-major f16, V transposed) ----
        for (int idx = tid; idx < BK * D_ / 4; idx += 128) {
            int kk = idx >> 4, d4 = (idx & 15) * 4;
            const float4 k4 = *(const float4*)(Kg + (size_t)(k0 + kk) * D_ + d4);
            const float4 w4 = *(const float4*)(Vg + (size_t)(k0 + kk) * D_ + d4);
            v4h hk = { (_Float16)k4.x, (_Float16)k4.y, (_Float16)k4.z, (_Float16)k4.w };
            *(v4h*)(&Ksm[kk * KSTR + d4]) = hk;              // ds_store_b64
            Vt[(d4 + 0) * VSTR + kk] = (_Float16)w4.x;
            Vt[(d4 + 1) * VSTR + kk] = (_Float16)w4.y;
            Vt[(d4 + 2) * VSTR + kk] = (_Float16)w4.z;
            Vt[(d4 + 3) * VSTR + kk] = (_Float16)w4.w;
        }
        if (k0 + BK < T_) {           // -> global_prefetch_b8
            __builtin_prefetch(Kg + (size_t)(k0 + BK) * D_ + tid * 32, 0, 3);
            __builtin_prefetch(Vg + (size_t)(k0 + BK) * D_ + tid * 32, 0, 3);
        }
        if (wave == 0) __builtin_amdgcn_s_wait_tensorcnt(0);  // TDM done -> LDS valid
        __syncthreads();

        // ---- S = Q K^T : 4 subtiles of 16 keys, K-dim = 64 (2 WMMA each) ----
        v8f s[4];
        #pragma unroll
        for (int sub = 0; sub < 4; ++sub) {
            v8f c = {};
            #pragma unroll
            for (int f = 0; f < 2; ++f) {
                // B fragment: col N = key = sub*16+ln ; half i<8 <-> d = 32f + 8*hl + i, i>=8 <-> d = 32f + 16 + 8*hl + (i-8)
                v16h b;
                const _Float16* bp = &Ksm[(sub * 16 + ln) * KSTR + 32 * f];
                #pragma unroll
                for (int i = 0; i < 8; ++i) { b[i] = bp[hl * 8 + i]; b[i + 8] = bp[16 + hl * 8 + i]; }
                c = __builtin_amdgcn_wmma_f32_16x16x32_f16(false, qa[f], false, b,
                                                           (short)0, c, false, false);
            }
            s[sub] = c;
        }

        // ---- mask (from LDS tile) + scale, online softmax over this block ----
        float rmax[8], rsum[8];
        #pragma unroll
        for (int r = 0; r < 8; ++r) rmax[r] = -3.0e38f;
        #pragma unroll
        for (int sub = 0; sub < 4; ++sub) {
            #pragma unroll
            for (int r = 0; r < 8; ++r) {
                int trow = wave * 16 + r + 8 * hl;           // row within mask tile
                int tcol = sub * 16 + ln;                    // col within mask tile
                bool m  = Msm[trow * BK + tcol] != 0;
                float sv = m ? -10000.0f : s[sub][r] * scale;
                s[sub][r] = sv;
                rmax[r] = fmaxf(rmax[r], sv);
            }
        }
        // row reduce across the 16 lanes of each half-wave (xor masks stay in-group)
        #pragma unroll
        for (int r = 0; r < 8; ++r) {
            float v = rmax[r];
            v = fmaxf(v, __shfl_xor(v, 1, 32));
            v = fmaxf(v, __shfl_xor(v, 2, 32));
            v = fmaxf(v, __shfl_xor(v, 4, 32));
            v = fmaxf(v, __shfl_xor(v, 8, 32));
            rmax[r] = v;
        }
        float alpha[8];
        #pragma unroll
        for (int r = 0; r < 8; ++r) {
            float mnew = fmaxf(mrun[r], rmax[r]);
            alpha[r] = __expf(mrun[r] - mnew);
            mrun[r]  = mnew;
            rsum[r]  = 0.0f;
        }
        #pragma unroll
        for (int sub = 0; sub < 4; ++sub) {
            #pragma unroll
            for (int r = 0; r < 8; ++r) {
                float pv = __expf(s[sub][r] - mrun[r]);  // masked -> exact 0
                s[sub][r] = pv;
                rsum[r] += pv;
            }
        }
        #pragma unroll
        for (int r = 0; r < 8; ++r) {
            float v = rsum[r];
            v += __shfl_xor(v, 1, 32);
            v += __shfl_xor(v, 2, 32);
            v += __shfl_xor(v, 4, 32);
            v += __shfl_xor(v, 8, 32);
            lrun[r] = lrun[r] * alpha[r] + v;
        }
        #pragma unroll
        for (int dt = 0; dt < 4; ++dt)
            #pragma unroll
            for (int r = 0; r < 8; ++r)
                acc[dt][r] *= alpha[r];

        // ---- P: C layout -> LDS -> A fragments (per-wave private region) ----
        _Float16* pw = &Psm[wave * 16 * PSTR];
        #pragma unroll
        for (int sub = 0; sub < 4; ++sub)
            #pragma unroll
            for (int r = 0; r < 8; ++r)
                pw[(r + 8 * hl) * PSTR + sub * 16 + ln] = (_Float16)s[sub][r];

        asm volatile("s_wait_dscnt 0x0" ::: "memory");

        v16h pa[2];
        #pragma unroll
        for (int g = 0; g < 2; ++g) {          // key offset = 32*g
            const _Float16* pp = &pw[ln * PSTR + 32 * g];
            #pragma unroll
            for (int i = 0; i < 8; ++i) {
                pa[g][i]     = pp[     hl * 8 + i];
                pa[g][i + 8] = pp[16 + hl * 8 + i];
            }
        }

        // ---- O += P V : 4 d-tiles x (K-dim 64 as 2 WMMA) ----
        #pragma unroll
        for (int dt = 0; dt < 4; ++dt) {
            v8f c = acc[dt];
            #pragma unroll
            for (int g = 0; g < 2; ++g) {
                // B fragment: col N = d = dt*16+ln ; half i<8 <-> key = 32g + 8*hl + i, i>=8 <-> key = 32g + 16 + 8*hl + (i-8)
                v16h b;
                const _Float16* bp = &Vt[(dt * 16 + ln) * VSTR + 32 * g];
                #pragma unroll
                for (int i = 0; i < 8; ++i) { b[i] = bp[hl * 8 + i]; b[i + 8] = bp[16 + hl * 8 + i]; }
                c = __builtin_amdgcn_wmma_f32_16x16x32_f16(false, pa[g], false, b,
                                                           (short)0, c, false, false);
            }
            acc[dt] = c;
        }
    }

    // ---- normalize rows and store ----
    float* Og = O + base;
    #pragma unroll
    for (int r = 0; r < 8; ++r) {
        int row = q0 + wave * 16 + r + 8 * hl;
        float inv = 1.0f / lrun[r];
        #pragma unroll
        for (int dt = 0; dt < 4; ++dt)
            Og[(size_t)row * D_ + dt * 16 + ln] = acc[dt][r] * inv;
    }
}

extern "C" void kernel_launch(void* const* d_in, const int* in_sizes, int n_in,
                              void* d_out, int out_size, void* d_ws, size_t ws_size,
                              hipStream_t stream) {
    const float* q = (const float*)d_in[0];
    const float* k = (const float*)d_in[1];
    const float* v = (const float*)d_in[2];
    const unsigned char* msk = (const unsigned char*)d_in[3];  // jnp.bool_ -> 1 byte
    float* out = (float*)d_out;

    dim3 grid(T_ / BM, B_ * H_);   // 32 query-blocks x 32 (b,h)
    dim3 block(128);               // 4 waves (wave32)
    fa_kernel<<<grid, block, 0, stream>>>(q, k, v, msk, out);
}
